// SimpleMamba3DBlock_87162066305521
// MI455X (gfx1250) — matrix operand order repaired and checked by reading, not compiled, not run on hardware
//
#include <hip/hip_runtime.h>
#include <hip/hip_bf16.h>

typedef __attribute__((ext_vector_type(16))) _Float16 v16h;
typedef __attribute__((ext_vector_type(8)))  float    v8f;

#define NB 16          // batch
#define SL 1024        // sequence length
#define DM 384         // d_model
#define DI 768         // d_inner
#define NS 16          // d_state
#define DTR 24         // dt_rank
#define MM (NB*SL)     // 16384 rows

// ---------------- WMMA fragment loaders (ISA 7.12.2 layouts) ----------------
// A (16x32 f16): lane row = lane&15; hi=lane>>4; halfs 0..7 -> K=kb+hi*8+i,
//                halfs 8..15 -> K=kb+16+hi*8+(i-8). Two 16B contiguous loads.
__device__ __forceinline__ v16h load_a_h(const _Float16* rowp, int kb, int hi) {
  union { v16h v; float4 q[2]; } u;
  u.q[0] = *reinterpret_cast<const float4*>(rowp + kb + hi * 8);
  u.q[1] = *reinterpret_cast<const float4*>(rowp + kb + 16 + hi * 8);
  return u.v;
}
// Same fragment but source data is f32 (converted in-register).
__device__ __forceinline__ v16h load_a_f(const float* rowp, int kb, int hi) {
  union { v16h v; _Float16 h[16]; } u;
  const float* p0 = rowp + kb + hi * 8;
  const float* p1 = rowp + kb + 16 + hi * 8;
#pragma unroll
  for (int i = 0; i < 8; ++i) { u.h[i] = (_Float16)p0[i]; u.h[8 + i] = (_Float16)p1[i]; }
  return u.v;
}
// B (32x16 f16): lane col = lane&15; hi=lane>>4; halfs 0..7 -> K=kb+hi*8+i,
// halfs 8..15 -> K=kb+16+hi*8+(i-8). Two 16B loads (weights stored (N,K) row-major).
__device__ __forceinline__ v16h load_b_h(const _Float16* rowp, int kb, int hi) {
  union { v16h v; float4 q[2]; } u;
  const _Float16* p = rowp + kb + hi * 8;
  u.q[0] = *reinterpret_cast<const float4*>(p);
  u.q[1] = *reinterpret_cast<const float4*>(p + 16);
  return u.v;
}

// ---------------- generic WMMA GEMM: out[m][n] = sum_k A[m][k]*W[n][k] ------
// Each wave computes a 16x64 output tile (1 A fragment reused across 4 B tiles).
// EPI: 0 = store f16 (ld=No)
//      1 = dbc split-store: n<24 -> dt16[m*32+n]; n<40 -> Bm; n<56 -> Cm;
//          n>=56 (zero-padded cols) -> dt16[m*32+n-32]  (zero-pads dt K=32)
//      2 = softplus(val + bias[n]) -> f16 (ld=No)
//      3 = resid[m*No+n] + val -> f32 out
template <int EPI, bool AFLOAT>
__global__ __launch_bounds__(256) void wmma_gemm(
    const void* __restrict__ Aptr, const _Float16* __restrict__ W,
    int Mo, int No, int K,
    _Float16* __restrict__ outh, float* __restrict__ outB, float* __restrict__ outC,
    const float* __restrict__ bias, const float* __restrict__ resid,
    float* __restrict__ outf)
{
  const int gw = (int)((blockIdx.x * blockDim.x + threadIdx.x) >> 5);
  const int lane = (int)(threadIdx.x & 31);
  const int ntiles = No >> 6;
  const int mt = gw / ntiles;
  const int nt = gw - mt * ntiles;
  if (mt >= (Mo >> 4)) return;                 // wave-uniform (EXEC stays full)
  const int row = lane & 15;
  const int hi = lane >> 4;

  const _Float16* Ah = (const _Float16*)Aptr + (size_t)(mt * 16 + row) * K;
  const float*    Af = (const float*)Aptr + (size_t)(mt * 16 + row) * K;
  const _Float16* w0 = W + (size_t)(nt * 64 +  0 + row) * K;
  const _Float16* w1 = W + (size_t)(nt * 64 + 16 + row) * K;
  const _Float16* w2 = W + (size_t)(nt * 64 + 32 + row) * K;
  const _Float16* w3 = W + (size_t)(nt * 64 + 48 + row) * K;

  v8f c0 = {}, c1 = {}, c2 = {}, c3 = {};
  for (int kb = 0; kb < K; kb += 32) {
    v16h a;
    if constexpr (AFLOAT) a = load_a_f(Af, kb, hi); else a = load_a_h(Ah, kb, hi);
    v16h b;
    b = load_b_h(w0, kb, hi);
    c0 = __builtin_amdgcn_wmma_f32_16x16x32_f16(false, a, false, b, (short)0, c0, false, false);
    b = load_b_h(w1, kb, hi);
    c1 = __builtin_amdgcn_wmma_f32_16x16x32_f16(false, a, false, b, (short)0, c1, false, false);
    b = load_b_h(w2, kb, hi);
    c2 = __builtin_amdgcn_wmma_f32_16x16x32_f16(false, a, false, b, (short)0, c2, false, false);
    b = load_b_h(w3, kb, hi);
    c3 = __builtin_amdgcn_wmma_f32_16x16x32_f16(false, a, false, b, (short)0, c3, false, false);
  }

  // C/D layout: lane col N=lane&15; element v -> M = v + (lane>>4)*8.
  v8f acc[4] = {c0, c1, c2, c3};
  const int m0 = mt * 16 + hi * 8;
  const int nbase = nt * 64 + row;
#pragma unroll
  for (int t = 0; t < 4; ++t) {
#pragma unroll
    for (int v = 0; v < 8; ++v) {
      float val = acc[t][v];
      int m = m0 + v;
      int n = nbase + t * 16;
      if constexpr (EPI == 0) {
        outh[(size_t)m * No + n] = (_Float16)val;
      } else if constexpr (EPI == 1) {
        if (n < 24)      outh[m * 32 + n] = (_Float16)val;
        else if (n < 40) outB[m * 16 + (n - 24)] = val;
        else if (n < 56) outC[m * 16 + (n - 40)] = val;
        else             outh[m * 32 + (n - 32)] = (_Float16)val;  // zeros
      } else if constexpr (EPI == 2) {
        float xv = val + bias[n];
        float sp = (xv > 20.f) ? xv : __logf(1.f + __expf(xv));
        outh[(size_t)m * No + n] = (_Float16)sp;
      } else {
        size_t o = (size_t)m * No + n;
        outf[o] = resid[o] + val;
      }
    }
  }
}

// ---------------- weight prep: f32 -> f16 (+ zero padding), A = -exp(Alog) --
__global__ __launch_bounds__(256) void prep_kernel(
    const float* __restrict__ in_proj, const float* __restrict__ xpf,
    const float* __restrict__ xpb, const float* __restrict__ dtwf,
    const float* __restrict__ dtwb, const float* __restrict__ outp,
    const float* __restrict__ Alf, const float* __restrict__ Alb,
    _Float16* __restrict__ wip, _Float16* __restrict__ wxpf,
    _Float16* __restrict__ wxpb, _Float16* __restrict__ wdtf,
    _Float16* __restrict__ wdtb, _Float16* __restrict__ wop,
    float* __restrict__ Anf, float* __restrict__ Anb)
{
  int i = (int)(blockIdx.x * blockDim.x + threadIdx.x);
  const int S0 = 1536 * 384, S1 = 64 * 768, S2 = 768 * 32, S3 = 384 * 768, S4 = 768 * 16;
  if (i < S0) { wip[i] = (_Float16)in_proj[i]; return; } i -= S0;
  if (i < S1) { int r = i / 768, c = i % 768; wxpf[i] = (_Float16)(r < 56 ? xpf[r * 768 + c] : 0.f); return; } i -= S1;
  if (i < S1) { int r = i / 768, c = i % 768; wxpb[i] = (_Float16)(r < 56 ? xpb[r * 768 + c] : 0.f); return; } i -= S1;
  if (i < S2) { int c = i % 32; wdtf[i] = (_Float16)(c < 24 ? dtwf[(i / 32) * 24 + c] : 0.f); return; } i -= S2;
  if (i < S2) { int c = i % 32; wdtb[i] = (_Float16)(c < 24 ? dtwb[(i / 32) * 24 + c] : 0.f); return; } i -= S2;
  if (i < S3) { wop[i] = (_Float16)outp[i]; return; } i -= S3;
  if (i < S4) { Anf[i] = -__expf(Alf[i]); return; } i -= S4;
  if (i < S4) { Anb[i] = -__expf(Alb[i]); }
}

// ---------------- LayerNorm: one wave32 per row of 384, f16 output ----------
__global__ __launch_bounds__(256) void ln_kernel(
    const float* __restrict__ x, const float* __restrict__ w,
    const float* __restrict__ b, _Float16* __restrict__ out)
{
  int wid = (int)((blockIdx.x * blockDim.x + threadIdx.x) >> 5);
  int lane = (int)(threadIdx.x & 31);
  if (wid >= MM) return;
  const float* rowp = x + (size_t)wid * DM;
  float v[12];
  float s = 0.f;
#pragma unroll
  for (int i = 0; i < 12; ++i) { v[i] = rowp[lane + i * 32]; s += v[i]; }
#pragma unroll
  for (int off = 16; off > 0; off >>= 1) s += __shfl_xor(s, off, 32);
  float mu = s * (1.f / DM);
  float var = 0.f;
#pragma unroll
  for (int i = 0; i < 12; ++i) { float d = v[i] - mu; var += d * d; }
#pragma unroll
  for (int off = 16; off > 0; off >>= 1) var += __shfl_xor(var, off, 32);
  float inv = rsqrtf(var * (1.f / DM) + 1e-5f);
#pragma unroll
  for (int i = 0; i < 12; ++i) {
    int c = lane + i * 32;
    out[(size_t)wid * DM + c] = (_Float16)((v[i] - mu) * inv * w[c] + b[c]);
  }
}

// ---------------- causal depthwise conv (dir-aware) + SiLU ------------------
// Logical time tau maps to physical p; fwd: p=tau (reads p-3..p),
// bwd (flipped seq): p=SL-1-tau (reads p..p+3). Unified: q = p - s*(3-k).
__global__ __launch_bounds__(256) void conv_silu_kernel(
    const _Float16* __restrict__ xz16, const float* __restrict__ cw,
    const float* __restrict__ cb, _Float16* __restrict__ xh16, int dir)
{
  int idx = (int)(blockIdx.x * blockDim.x + threadIdx.x);
  if (idx >= MM * DI) return;
  int dch = idx % DI;
  int p = (idx / DI) % SL;
  int bb = idx / (DI * SL);
  int s = (dir == 0) ? 1 : -1;
  float acc = cb[dch];
#pragma unroll
  for (int k = 0; k < 4; ++k) {
    int q = p - s * (3 - k);
    if (q >= 0 && q < SL)
      acc += cw[dch * 4 + k] * (float)xz16[((size_t)(bb * SL + q)) * 1536 + dch];
  }
  float sv = acc / (1.f + __expf(-acc));
  xh16[(size_t)(bb * SL + p) * DI + dch] = (_Float16)sv;
}

// ---------------- selective scan: one lane per (b, channel) -----------------
__global__ __launch_bounds__(256) void scan_kernel(
    const _Float16* __restrict__ delta16, const _Float16* __restrict__ xh16,
    const float* __restrict__ Bmat, const float* __restrict__ Cmat,
    const float* __restrict__ Aneg, const float* __restrict__ Dv,
    const _Float16* __restrict__ xz16, float* __restrict__ ysum, int dir)
{
  int idx = (int)(blockIdx.x * blockDim.x + threadIdx.x);
  if (idx >= NB * DI) return;
  int dch = idx % DI;
  int bb = idx / DI;
  float Ar[16];
#pragma unroll
  for (int n = 0; n < 16; ++n) Ar[n] = Aneg[dch * 16 + n];
  const float Dvv = Dv[dch];
  float h[16];
#pragma unroll
  for (int n = 0; n < 16; ++n) h[n] = 0.f;

  for (int t = 0; t < SL; ++t) {
    int p = (dir == 0) ? t : (SL - 1 - t);
    int base = bb * SL + p;
    float dt = (float)delta16[(size_t)base * DI + dch];
    float u  = (float)xh16[(size_t)base * DI + dch];
    const float4* Bp = (const float4*)(Bmat + (size_t)base * 16);
    const float4* Cp = (const float4*)(Cmat + (size_t)base * 16);
    float Bt[16], Ct[16];
#pragma unroll
    for (int q = 0; q < 4; ++q) {
      float4 bq = Bp[q], cq = Cp[q];
      Bt[q * 4 + 0] = bq.x; Bt[q * 4 + 1] = bq.y; Bt[q * 4 + 2] = bq.z; Bt[q * 4 + 3] = bq.w;
      Ct[q * 4 + 0] = cq.x; Ct[q * 4 + 1] = cq.y; Ct[q * 4 + 2] = cq.z; Ct[q * 4 + 3] = cq.w;
    }
    float du = dt * u;
    float y = 0.f;
#pragma unroll
    for (int n = 0; n < 16; ++n) {
      float dA = __expf(dt * Ar[n]);
      h[n] = h[n] * dA + du * Bt[n];
      y += h[n] * Ct[n];
    }
    float zz = (float)xz16[(size_t)base * 1536 + DI + dch];
    float outv = (y + u * Dvv) * (zz / (1.f + __expf(-zz)));
    size_t o = (size_t)base * DI + dch;
    if (dir == 0) ysum[o] = outv; else ysum[o] += outv;
  }
}

// ---------------------------------------------------------------------------
extern "C" void kernel_launch(void* const* d_in, const int* in_sizes, int n_in,
                              void* d_out, int out_size, void* d_ws, size_t ws_size,
                              hipStream_t stream)
{
  (void)in_sizes; (void)n_in; (void)out_size; (void)ws_size;
  const float* x        = (const float*)d_in[0];
  const float* ln_w     = (const float*)d_in[1];
  const float* ln_b     = (const float*)d_in[2];
  const float* in_projw = (const float*)d_in[3];
  const float* out_pw   = (const float*)d_in[4];
  const float* conv_w[2]  = {(const float*)d_in[5],  (const float*)d_in[12]};
  const float* conv_b[2]  = {(const float*)d_in[6],  (const float*)d_in[13]};
  const float* xproj_w[2] = {(const float*)d_in[7],  (const float*)d_in[14]};
  const float* dt_w[2]    = {(const float*)d_in[8],  (const float*)d_in[15]};
  const float* dt_b[2]    = {(const float*)d_in[9],  (const float*)d_in[16]};
  const float* A_log[2]   = {(const float*)d_in[10], (const float*)d_in[17]};
  const float* Dvec[2]    = {(const float*)d_in[11], (const float*)d_in[18]};

  char* p = (char*)d_ws;
  auto alloc = [&](size_t bytes) -> char* {
    char* r = p; p += (bytes + 255) & ~(size_t)255; return r;
  };
  _Float16* xn16    = (_Float16*)alloc((size_t)MM * DM * 2);
  _Float16* wip16   = (_Float16*)alloc((size_t)1536 * 384 * 2);
  _Float16* wxp16_f = (_Float16*)alloc((size_t)64 * 768 * 2);
  _Float16* wxp16_b = (_Float16*)alloc((size_t)64 * 768 * 2);
  _Float16* wdt16_f = (_Float16*)alloc((size_t)768 * 32 * 2);
  _Float16* wdt16_b = (_Float16*)alloc((size_t)768 * 32 * 2);
  _Float16* wop16   = (_Float16*)alloc((size_t)384 * 768 * 2);
  float*    Aneg_f  = (float*)alloc((size_t)768 * 16 * 4);
  float*    Aneg_b  = (float*)alloc((size_t)768 * 16 * 4);
  _Float16* xz16    = (_Float16*)alloc((size_t)MM * 1536 * 2);
  _Float16* xh16    = (_Float16*)alloc((size_t)MM * DI * 2);
  _Float16* dt16    = (_Float16*)alloc((size_t)MM * 32 * 2);
  float*    Bm      = (float*)alloc((size_t)MM * 16 * 4);
  float*    Cm      = (float*)alloc((size_t)MM * 16 * 4);
  _Float16* del16   = (_Float16*)alloc((size_t)MM * DI * 2);
  float*    ysum    = (float*)alloc((size_t)MM * DI * 4);
  _Float16* wxp16[2] = {wxp16_f, wxp16_b};
  _Float16* wdt16[2] = {wdt16_f, wdt16_b};
  float*    Aneg[2]  = {Aneg_f, Aneg_b};

  // 1) weight prep (f32->f16 + padding, A = -exp(A_log))
  {
    const int total = 1536 * 384 + 2 * 64 * 768 + 2 * 768 * 32 + 384 * 768 + 2 * 768 * 16;
    prep_kernel<<<(total + 255) / 256, 256, 0, stream>>>(
        in_projw, xproj_w[0], xproj_w[1], dt_w[0], dt_w[1], out_pw, A_log[0], A_log[1],
        wip16, wxp16_f, wxp16_b, wdt16_f, wdt16_b, wop16, Aneg_f, Aneg_b);
  }
  // 2) LayerNorm -> xn16
  ln_kernel<<<(MM * 32) / 256, 256, 0, stream>>>(x, ln_w, ln_b, xn16);
  // 3) in_proj GEMM: (16384x384) x (1536x384)^T -> xz16 f16
  {
    int waves = (MM / 16) * (1536 / 64);
    wmma_gemm<0, false><<<waves * 32 / 256, 256, 0, stream>>>(
        xn16, wip16, MM, 1536, 384, xz16, nullptr, nullptr, nullptr, nullptr, nullptr);
  }
  // 4) per-direction branch
  for (int dir = 0; dir < 2; ++dir) {
    conv_silu_kernel<<<(MM * DI) / 256, 256, 0, stream>>>(
        xz16, conv_w[dir], conv_b[dir], xh16, dir);
    // x_proj GEMM: K=768, N=64 (padded); split-store dt16 / Bm / Cm
    {
      int waves = (MM / 16) * (64 / 64);
      wmma_gemm<1, false><<<waves * 32 / 256, 256, 0, stream>>>(
          xh16, wxp16[dir], MM, 64, DI, dt16, Bm, Cm, nullptr, nullptr, nullptr);
    }
    // dt_proj GEMM: K=32 (padded), N=768; softplus(+dt_b) -> del16
    {
      int waves = (MM / 16) * (DI / 64);
      wmma_gemm<2, false><<<waves * 32 / 256, 256, 0, stream>>>(
          dt16, wdt16[dir], MM, DI, 32, del16, nullptr, nullptr, dt_b[dir], nullptr, nullptr);
    }
    // selective scan + *silu(z); fwd stores, bwd accumulates into ysum
    scan_kernel<<<(NB * DI) / 256, 256, 0, stream>>>(
        del16, xh16, Bm, Cm, Aneg[dir], Dvec[dir], xz16, ysum, dir);
  }
  // 5) out_proj GEMM (A is f32 ysum, converted in fragment loader) + residual
  {
    int waves = (MM / 16) * (DM / 64);
    wmma_gemm<3, true><<<waves * 32 / 256, 256, 0, stream>>>(
        ysum, wop16, MM, DM, DI, nullptr, nullptr, nullptr, nullptr, x, (float*)d_out);
  }
}
